// LFMMILoss2_47038481826031
// MI455X (gfx1250) — matrix operand order repaired and checked by reading, not compiled, not run on hardware
//
#include <hip/hip_runtime.h>
#include <hip/hip_bf16.h>
#include <math.h>

#define Bn 32
#define Tn 400
#define Cn 3000
#define SNs 128
#define SDs 512
#define NEGC (-1e4f)

typedef __attribute__((ext_vector_type(16))) _Float16 v16h;
typedef __attribute__((ext_vector_type(8)))  _Float16 v8h;
typedef __attribute__((ext_vector_type(8)))  float    v8f;

__device__ __forceinline__ float clip30(float v) {
  return fminf(30.f, fmaxf(-30.f, v));
}

// ---------------------------------------------------------------------------
// prep: ET[s][i] = exp(den_trans[i][s])  (f16, transposed for contiguous
// WMMA B-fragment loads: lane reads 16 consecutive K values at fixed N)
// ---------------------------------------------------------------------------
__global__ void prep_ET(const float* __restrict__ dt, _Float16* __restrict__ ET) {
  int idx = blockIdx.x * blockDim.x + threadIdx.x;
  if (idx >= SDs * SDs) return;
  int s = idx / SDs, i = idx - s * SDs;
  ET[s * SDs + i] = (_Float16)expf(dt[i * SDs + s]);
}

// ---------------------------------------------------------------------------
// Numerator: banded left-to-right chain. 1 block per batch, 1 thread/state.
// Off-band (-1e4) terms underflow in f32 logsumexp, so the 2-term band is
// exact to f32 precision.
// ---------------------------------------------------------------------------
__global__ void num_fwd(const float* __restrict__ x, const int* __restrict__ seqlen,
                        const float* __restrict__ ntrans, const int* __restrict__ npdf,
                        const float* __restrict__ ninit, const float* __restrict__ nfinal,
                        float* __restrict__ num_llh) {
  __shared__ float alpha[SNs];
  __shared__ float red[SNs];
  const int b = blockIdx.x, s = threadIdx.x;
  const int pdf = npdf[b * SNs + s];
  const float* xb = x + (size_t)b * Tn * Cn;
  const float t_self = ntrans[((size_t)b * SNs + s) * SNs + s];
  const float t_prev = (s > 0) ? ntrans[((size_t)b * SNs + (s - 1)) * SNs + s] : NEGC;
  alpha[s] = ninit[b * SNs + s] + clip30(xb[pdf]);
  const int L = seqlen[b];
  __syncthreads();
  for (int t = 1; t < L; ++t) {
    float as = alpha[s] + t_self;
    float ap = (s > 0 ? alpha[s - 1] : -3.0e38f) + t_prev;
    float m = fmaxf(as, ap);
    float v = m + logf(expf(as - m) + expf(ap - m));
    float em = clip30(xb[(size_t)t * Cn + pdf]);
    __syncthreads();
    alpha[s] = v + em;
    __syncthreads();
  }
  // logsumexp over states of alpha + final
  float val = alpha[s] + nfinal[b * SNs + s];
  red[s] = val;
  __syncthreads();
  for (int o = 64; o > 0; o >>= 1) { if (s < o) red[s] = fmaxf(red[s], red[s + o]); __syncthreads(); }
  float m = red[0];
  __syncthreads();
  red[s] = expf(val - m);
  __syncthreads();
  for (int o = 64; o > 0; o >>= 1) { if (s < o) red[s] += red[s + o]; __syncthreads(); }
  if (s == 0) num_llh[b] = m + logf(red[0]);
}

// ---------------------------------------------------------------------------
// Denominator: 2 blocks x 16 batches, 512 threads = 16 waves.
// Per step: wave w computes m/P for its row; then each wave does a
// 16x512 @ 512x32 f16 WMMA GEMM slice (2 N-tiles, K=512 in 16 chunks of 32)
// and folds m + log(acc) + emission straight back into LDS alpha.
// ---------------------------------------------------------------------------
__global__ void __launch_bounds__(512) den_fwd(
    const float* __restrict__ x, const int* __restrict__ seqlen,
    const _Float16* __restrict__ ET, const int* __restrict__ dpdf,
    const float* __restrict__ dinit, const float* __restrict__ dfinal,
    float* __restrict__ den_llh) {
  __shared__ __align__(16) float    alpha[16][SDs];   // 32 KB
  __shared__ __align__(16) _Float16 P[16][SDs];       // 16 KB
  __shared__ float m_lds[16];
  __shared__ int   L_lds[16];
  __shared__ int   maxL_s;

  const int tid  = threadIdx.x;
  const int w    = tid >> 5;        // wave id == local batch row, and N-tile pair owner
  const int lane = tid & 31;
  const int b0   = blockIdx.x * 16;
  const int b    = b0 + w;

  // GEMM output columns owned by this lane (two 16-wide N tiles per wave)
  const int n0 = (2 * w) * 16 + (lane & 15);
  const int n1 = (2 * w + 1) * 16 + (lane & 15);
  const int pdf_n0 = dpdf[n0];
  const int pdf_n1 = dpdf[n1];

  if (lane == 0) L_lds[w] = seqlen[b];
  __syncthreads();
  if (tid == 0) {
    int mx = 0;
    for (int i = 0; i < 16; ++i) mx = max(mx, L_lds[i]);
    maxL_s = mx;
  }
  // t = 0 init for own row
  const float* xb = x + (size_t)b * Tn * Cn;
#pragma unroll
  for (int j = 0; j < 16; ++j) {
    int s = lane + 32 * j;
    alpha[w][s] = dinit[s] + clip30(xb[dpdf[s]]);
  }
  __syncthreads();
  const int maxL = maxL_s;

  for (int t = 1; t < maxL; ++t) {
    // ---- phase 1: active rows -> row max m, P = exp(alpha - m) in f16 ----
    if (t < L_lds[w]) {
      float a[16], mloc = -3.0e38f;
#pragma unroll
      for (int j = 0; j < 16; ++j) { a[j] = alpha[w][lane + 32 * j]; mloc = fmaxf(mloc, a[j]); }
      for (int o = 16; o > 0; o >>= 1) mloc = fmaxf(mloc, __shfl_xor(mloc, o, 32));
#pragma unroll
      for (int j = 0; j < 16; ++j) P[w][lane + 32 * j] = (_Float16)expf(a[j] - mloc);
      if (lane == 0) m_lds[w] = mloc;
    }
    __syncthreads();

    // ---- phase 2: WMMA GEMM + in-place alpha update --------------------
    v8f c0 = {}, c1 = {};
    const int arow = lane & 15;
    const int koff = (lane < 16) ? 0 : 16;  // A-fragment K sub-offset
    const int kb16 = (lane < 16) ? 0 : 16;  // B-fragment K sub-offset
#pragma unroll 4
    for (int kb = 0; kb < 16; ++kb) {
      const int kbase = kb * 32;
      union { v16h v; v8h h[2]; } af;
      af.h[0] = *reinterpret_cast<const v8h*>(&P[arow][kbase + koff]);
      af.h[1] = *reinterpret_cast<const v8h*>(&P[arow][kbase + koff + 8]);
      v16h bf0 = *reinterpret_cast<const v16h*>(&ET[(size_t)n0 * SDs + kbase + kb16]);
      v16h bf1 = *reinterpret_cast<const v16h*>(&ET[(size_t)n1 * SDs + kbase + kb16]);
      c0 = __builtin_amdgcn_wmma_f32_16x16x32_f16(false, af.v, false, bf0, (short)0, c0, false, false);
      c1 = __builtin_amdgcn_wmma_f32_16x16x32_f16(false, af.v, false, bf1, (short)0, c1, false, false);
    }
    // D layout: element r -> M = r + 8*(lane>=16), N = lane%16 (within tile)
    const int rbase = (lane < 16) ? 0 : 8;
#pragma unroll
    for (int r = 0; r < 8; ++r) {
      const int row = rbase + r;
      if (t < L_lds[row]) {
        const float mrow = m_lds[row];
        const float* xr = x + ((size_t)(b0 + row) * Tn + t) * Cn;
        alpha[row][n0] = mrow + logf(c0[r]) + clip30(xr[pdf_n0]);
        alpha[row][n1] = mrow + logf(c1[r]) + clip30(xr[pdf_n1]);
      }
    }
    __syncthreads();
  }

  // ---- final logsumexp(alpha + final) per row -------------------------
  float vals[16], mloc = -3.0e38f;
#pragma unroll
  for (int j = 0; j < 16; ++j) {
    int s = lane + 32 * j;
    vals[j] = alpha[w][s] + dfinal[s];
    mloc = fmaxf(mloc, vals[j]);
  }
  for (int o = 16; o > 0; o >>= 1) mloc = fmaxf(mloc, __shfl_xor(mloc, o, 32));
  float sum = 0.f;
#pragma unroll
  for (int j = 0; j < 16; ++j) sum += expf(vals[j] - mloc);
  for (int o = 16; o > 0; o >>= 1) sum += __shfl_xor(sum, o, 32);
  if (lane == 0) den_llh[b] = mloc + logf(sum);
}

// ---------------------------------------------------------------------------
__global__ void finalize_loss(const float* __restrict__ num_llh,
                              const float* __restrict__ den_llh,
                              float* __restrict__ out) {
  const int lane = threadIdx.x;
  float v = num_llh[lane] - den_llh[lane];
  for (int o = 16; o > 0; o >>= 1) v += __shfl_xor(v, o, 32);
  if (lane == 0) out[0] = -v;
}

extern "C" void kernel_launch(void* const* d_in, const int* in_sizes, int n_in,
                              void* d_out, int out_size, void* d_ws, size_t ws_size,
                              hipStream_t stream) {
  (void)in_sizes; (void)n_in; (void)out_size; (void)ws_size;
  const float* x      = (const float*)d_in[0];
  const int*   seqlen = (const int*)d_in[1];
  const float* ntrans = (const float*)d_in[2];
  const int*   npdf   = (const int*)d_in[3];
  const float* ninit  = (const float*)d_in[4];
  const float* nfinal = (const float*)d_in[5];
  const float* dtrans = (const float*)d_in[6];
  const int*   dpdf   = (const int*)d_in[7];
  const float* dinit  = (const float*)d_in[8];
  const float* dfinal = (const float*)d_in[9];
  float* out = (float*)d_out;

  _Float16* ET    = (_Float16*)d_ws;
  float* num_llh  = (float*)((char*)d_ws + (size_t)SDs * SDs * sizeof(_Float16));
  float* den_llh  = num_llh + Bn;

  prep_ET<<<(SDs * SDs + 255) / 256, 256, 0, stream>>>(dtrans, ET);
  num_fwd<<<Bn, SNs, 0, stream>>>(x, seqlen, ntrans, npdf, ninit, nfinal, num_llh);
  den_fwd<<<2, 512, 0, stream>>>(x, seqlen, ET, dpdf, dinit, dfinal, den_llh);
  finalize_loss<<<1, 32, 0, stream>>>(num_llh, den_llh, out);
}
